// Head_52286931861726
// MI455X (gfx1250) — matrix operand order repaired and checked by reading, not compiled, not run on hardware
//
#include <hip/hip_runtime.h>

// ---------------------------------------------------------------------------
// Single-head causal attention for MI455X (gfx1250), bf16 WMMA pipeline with
// async (ASYNCcnt) K/V staging into LDS, double-buffered.
// B=4, T=4096, E=1024, H=128.  All matmuls via v_wmma_f32_16x16x32_bf16.
// ---------------------------------------------------------------------------

typedef __bf16 bf16_t;
typedef __attribute__((ext_vector_type(4)))  __bf16 v4bf;
typedef __attribute__((ext_vector_type(8)))  __bf16 v8bf;
typedef __attribute__((ext_vector_type(16))) __bf16 v16bf;
typedef __attribute__((ext_vector_type(8)))  float  v8f;

constexpr int Bn = 4;
constexpr int Tn = 4096;
constexpr int En = 1024;
constexpr int Hn = 128;
constexpr int Mn = Bn * Tn;                 // 16384 flattened rows
constexpr float ATTN_SCALE = 0.08838834764831845f;  // 1/sqrt(128)

__device__ __forceinline__ v16bf cat16(v8bf lo, v8bf hi) {
  return __builtin_shufflevector(lo, hi, 0,1,2,3,4,5,6,7,8,9,10,11,12,13,14,15);
}

__device__ __forceinline__ v8f wmma_bf16(v16bf a, v16bf b, v8f c) {
  // 8 args: (neg_a, A, neg_b, B, c_mod, C, reuse_a, reuse_b)
  return __builtin_amdgcn_wmma_f32_16x16x32_bf16(false, a, false, b, (short)0, c,
                                                 false, false);
}

// Async global->LDS copy of one 16B chunk (GV mode), tracked by ASYNCcnt.
__device__ __forceinline__ void async_b128(unsigned lds_addr, const void* gaddr) {
  asm volatile("global_load_async_to_lds_b128 %0, %1, off"
               :: "v"(lds_addr), "v"(gaddr) : "memory");
}
__device__ __forceinline__ unsigned lds_off(const void* p) {
  return (unsigned)(unsigned long long)p;   // low 32 bits of generic = LDS offset
}

// ---------------------------------------------------------------------------
// fp32 -> bf16 conversion (vectorized: 4 elements / thread)
// ---------------------------------------------------------------------------
__global__ __launch_bounds__(256)
void cvt_f32_bf16(const float* __restrict__ in, bf16_t* __restrict__ out, int n4) {
  int i = blockIdx.x * blockDim.x + threadIdx.x;
  if (i < n4) {
    float4 v = ((const float4*)in)[i];
    v4bf o;
    o[0] = (bf16_t)v.x; o[1] = (bf16_t)v.y; o[2] = (bf16_t)v.z; o[3] = (bf16_t)v.w;
    ((v4bf*)out)[i] = o;
  }
}

// ---------------------------------------------------------------------------
// QKV projection: y[m, h] = sum_e x[m, e] * W[h, e]
//   blockIdx.x -> 16-row tile of M (=B*T), blockIdx.y -> matrix (0=Q,1=K,2=V)
//   8 waves per block, each wave: one 16x16 output tile (cols = wave*16..+15)
//   V is stored transposed [Hn][Mn] so the attention PV B-fragment is contiguous.
// ---------------------------------------------------------------------------
__global__ __launch_bounds__(256)
void qkv_gemm(const bf16_t* __restrict__ xb, const bf16_t* __restrict__ wb,
              bf16_t* __restrict__ qb, bf16_t* __restrict__ kbuf,
              bf16_t* __restrict__ vtb) {
  const int lane = threadIdx.x & 31;
  const int wv   = threadIdx.x >> 5;
  const int hi   = lane >> 4;        // half-wave select
  const int ln   = lane & 15;
  const int m    = blockIdx.y;       // 0=Q 1=K 2=V
  const int rowbase = blockIdx.x * 16;
  const int colbase = wv * 16;

  const bf16_t* __restrict__ W    = wb + (size_t)m * Hn * En;
  const bf16_t* __restrict__ arow = xb + (size_t)(rowbase + ln) * En;
  const bf16_t* __restrict__ brow = W  + (size_t)(colbase + ln) * En;

  v8f acc = {};
#pragma unroll 4
  for (int k0 = 0; k0 < En; k0 += 32) {
    // A fragment: 16x32 bf16, per ISA layout (chunks at k0+hi*8 and +16)
    v8bf alo = *(const v8bf*)(arow + k0 + hi * 8);
    v8bf ahi = *(const v8bf*)(arow + k0 + hi * 8 + 16);
    // B fragment: 32x16 bf16, column = W row, chunks at k0+hi*8 and +16
    v8bf blo = *(const v8bf*)(brow + k0 + hi * 8);
    v8bf bhi = *(const v8bf*)(brow + k0 + hi * 8 + 16);
    acc = wmma_bf16(cat16(alo, ahi), cat16(blo, bhi), acc);
  }

  const int col = colbase + ln;
  if (m == 2) {
    // V transposed: vtb[col][row]
#pragma unroll
    for (int j = 0; j < 8; ++j) {
      int row = rowbase + j + hi * 8;
      vtb[(size_t)col * Mn + row] = (bf16_t)acc[j];
    }
  } else {
    bf16_t* __restrict__ dst = (m == 0) ? qb : kbuf;
#pragma unroll
    for (int j = 0; j < 8; ++j) {
      int row = rowbase + j + hi * 8;
      dst[(size_t)row * Hn + col] = (bf16_t)acc[j];
    }
  }
}

// ---------------------------------------------------------------------------
// Flash attention (causal, online softmax), block-cooperative K/V staging.
//   Block = 8 waves = 128 consecutive query rows of one batch; all waves walk
//   the same key range (masked tail tiles for the lower waves are cheap and
//   numerically inert).  K/V tiles are DMA'd into LDS with
//   global_load_async_to_lds_b128 (double-buffered, one barrier per tile) and
//   consumed via ds_load_b128 WMMA fragments.
// ---------------------------------------------------------------------------
__global__ __launch_bounds__(256)
void flash_attn(const bf16_t* __restrict__ qb, const bf16_t* __restrict__ kbuf,
                const bf16_t* __restrict__ vtb, float* __restrict__ out) {
  __shared__ bf16_t ktile[2][32][128];   // 2 x 8KB : 32 keys x 128 dims
  __shared__ bf16_t vtile[2][128][32];   // 2 x 8KB : 128 dims x 32 keys
  __shared__ bf16_t pbuf[8][16][40];     // P transpose, stride 40 keeps 16B align

  const int tid  = threadIdx.x;
  const int lane = tid & 31;
  const int wv   = tid >> 5;
  const int hi   = lane >> 4;
  const int ln   = lane & 15;

  const int b     = blockIdx.x / (Tn / 128);
  const int qb0   = (blockIdx.x % (Tn / 128)) * 128;   // block's first query row
  const int qbase = qb0 + wv * 16;                     // this wave's first row
  const int nkt   = (qb0 + 128) / 32;                  // # of 32-key tiles

  // ---- async tile issue: K tile is one contiguous 8KB run; V^T tile is
  //      128 rows x 64B.  512 chunks of 16B each -> 2 per thread per tensor.
  auto issue_tile = [&](int kt, int sel) {
    const char* kg = (const char*)(kbuf + (size_t)(b * Tn + kt * 32) * Hn);
    unsigned    kl = lds_off(&ktile[sel][0][0]);
#pragma unroll
    for (int c = 0; c < 2; ++c) {
      int chunk = tid + c * 256;
      async_b128(kl + chunk * 16, kg + chunk * 16);
    }
    unsigned vl = lds_off(&vtile[sel][0][0]);
#pragma unroll
    for (int c = 0; c < 2; ++c) {
      int chunk = tid + c * 256;          // 0..511
      int row   = chunk >> 2;             // head column 0..127
      int q4    = chunk & 3;              // 16B sub-chunk of the 64B row
      const char* g = (const char*)(vtb + (size_t)row * Mn + (size_t)b * Tn + kt * 32)
                      + q4 * 16;
      async_b128(vl + chunk * 16, g);
    }
  };

  // ---- Q fragments (reused across whole key loop): 4 k-steps over H=128
  const bf16_t* __restrict__ qrow = qb + (size_t)(b * Tn + qbase + ln) * Hn;
  v16bf qf[4];
#pragma unroll
  for (int kk = 0; kk < 4; ++kk) {
    v8bf lo  = *(const v8bf*)(qrow + kk * 32 + hi * 8);
    v8bf hi8 = *(const v8bf*)(qrow + kk * 32 + hi * 8 + 16);
    qf[kk] = cat16(lo, hi8);
  }

  v8f acc[8];
#pragma unroll
  for (int c = 0; c < 8; ++c) acc[c] = v8f{};
  float mi[8], li[8];
#pragma unroll
  for (int j = 0; j < 8; ++j) { mi[j] = -__builtin_inff(); li[j] = 0.0f; }

  issue_tile(0, 0);                       // prime the pipeline

  for (int kt = 0; kt < nkt; ++kt) {
    const int kbp = kt * 32;
    const int sel = kt & 1;

    asm volatile("s_wait_asynccnt 0x0" ::: "memory");  // my chunks of tile kt landed
    __syncthreads();             // everyone's chunks landed; other buffer is free
    if (kt + 1 < nkt) issue_tile(kt + 1, sel ^ 1);     // overlap next DMA

    // ---- S = Q * K^T, two 16x16 column halves, K fragments from LDS -------
    v8f s0 = {}, s1 = {};
#pragma unroll
    for (int kk = 0; kk < 4; ++kk) {
      v8bf k0lo = *(const v8bf*)(&ktile[sel][ln][kk * 32 + hi * 8]);
      v8bf k0hi = *(const v8bf*)(&ktile[sel][ln][kk * 32 + hi * 8 + 16]);
      s0 = wmma_bf16(qf[kk], cat16(k0lo, k0hi), s0);
      v8bf k1lo = *(const v8bf*)(&ktile[sel][16 + ln][kk * 32 + hi * 8]);
      v8bf k1hi = *(const v8bf*)(&ktile[sel][16 + ln][kk * 32 + hi * 8 + 16]);
      s1 = wmma_bf16(qf[kk], cat16(k1lo, k1hi), s1);
    }

    // ---- scale + causal mask + online softmax -----------------------------
    const int c0 = kbp + ln;
    const int c1 = c0 + 16;
    float mx[8];
#pragma unroll
    for (int j = 0; j < 8; ++j) {
      int rg = qbase + j + hi * 8;                  // global query row
      float a0 = s0[j] * ATTN_SCALE;
      float a1 = s1[j] * ATTN_SCALE;
      a0 = (c0 > rg) ? -__builtin_inff() : a0;       // cndmask, no divergence
      a1 = (c1 > rg) ? -__builtin_inff() : a1;
      s0[j] = a0; s1[j] = a1;
      mx[j] = fmaxf(a0, a1);
    }
#pragma unroll
    for (int msk = 1; msk <= 8; msk <<= 1)
#pragma unroll
      for (int j = 0; j < 8; ++j)
        mx[j] = fmaxf(mx[j], __shfl_xor(mx[j], msk, 32));

    float al[8], rs[8];
#pragma unroll
    for (int j = 0; j < 8; ++j) {
      float mn = fmaxf(mi[j], mx[j]);
      al[j] = __expf(mi[j] - mn);
      mi[j] = mn;
      float p0 = __expf(s0[j] - mn);
      float p1 = __expf(s1[j] - mn);
      s0[j] = p0; s1[j] = p1;
      rs[j] = p0 + p1;
    }
#pragma unroll
    for (int msk = 1; msk <= 8; msk <<= 1)
#pragma unroll
      for (int j = 0; j < 8; ++j)
        rs[j] += __shfl_xor(rs[j], msk, 32);
#pragma unroll
    for (int j = 0; j < 8; ++j) li[j] = li[j] * al[j] + rs[j];
#pragma unroll
    for (int c = 0; c < 8; ++c)
#pragma unroll
      for (int j = 0; j < 8; ++j) acc[c][j] *= al[j];

    // ---- P (C/D layout, f32) -> LDS row-major bf16 -> A-fragment ----------
#pragma unroll
    for (int j = 0; j < 8; ++j) {
      int r = j + hi * 8;
      pbuf[wv][r][ln]      = (bf16_t)s0[j];
      pbuf[wv][r][16 + ln] = (bf16_t)s1[j];
    }
    asm volatile("s_wait_dscnt 0x0" ::: "memory");   // cross-lane LDS dep (intra-wave)
    v8bf plo = *(const v8bf*)(&pbuf[wv][ln][hi * 8]);
    v8bf phi = *(const v8bf*)(&pbuf[wv][ln][hi * 8 + 16]);
    v16bf pf = cat16(plo, phi);

    // ---- O += P * V, V fragments from LDS (row = head col, keys contiguous)
#pragma unroll
    for (int c = 0; c < 8; ++c) {
      v8bf vlo = *(const v8bf*)(&vtile[sel][c * 16 + ln][hi * 8]);
      v8bf vhi = *(const v8bf*)(&vtile[sel][c * 16 + ln][hi * 8 + 16]);
      acc[c] = wmma_bf16(pf, cat16(vlo, vhi), acc[c]);
    }
  }

  // ---- epilogue: normalize and store fp32 output --------------------------
  float inv[8];
#pragma unroll
  for (int j = 0; j < 8; ++j) inv[j] = 1.0f / li[j];
#pragma unroll
  for (int c = 0; c < 8; ++c) {
#pragma unroll
    for (int j = 0; j < 8; ++j) {
      int row = b * Tn + qbase + j + hi * 8;
      out[(size_t)row * Hn + c * 16 + ln] = acc[c][j] * inv[j];
    }
  }
}

// ---------------------------------------------------------------------------
// Launch
// ---------------------------------------------------------------------------
extern "C" void kernel_launch(void* const* d_in, const int* in_sizes, int n_in,
                              void* d_out, int out_size, void* d_ws, size_t ws_size,
                              hipStream_t stream) {
  const float* x  = (const float*)d_in[0];   // [B,T,E]
  const float* Wq = (const float*)d_in[1];   // [H,E]
  const float* Wk = (const float*)d_in[2];
  const float* Wv = (const float*)d_in[3];
  float* out = (float*)d_out;                // [B,T,H]

  // workspace layout (bf16): x | W(q,k,v) | Q | K | V^T  (~47 MB)
  char* ws = (char*)d_ws;
  size_t off = 0;
  bf16_t* xb   = (bf16_t*)(ws + off); off += (size_t)Mn * En * sizeof(bf16_t);
  bf16_t* wb   = (bf16_t*)(ws + off); off += (size_t)3 * Hn * En * sizeof(bf16_t);
  bf16_t* qb   = (bf16_t*)(ws + off); off += (size_t)Mn * Hn * sizeof(bf16_t);
  bf16_t* kbuf = (bf16_t*)(ws + off); off += (size_t)Mn * Hn * sizeof(bf16_t);
  bf16_t* vtb  = (bf16_t*)(ws + off);

  // 1) fp32 -> bf16 conversions
  {
    int n4 = Mn * En / 4;
    cvt_f32_bf16<<<(n4 + 255) / 256, 256, 0, stream>>>(x, xb, n4);
    int w4 = Hn * En / 4;
    cvt_f32_bf16<<<(w4 + 255) / 256, 256, 0, stream>>>(Wq, wb + 0 * (size_t)Hn * En, w4);
    cvt_f32_bf16<<<(w4 + 255) / 256, 256, 0, stream>>>(Wk, wb + 1 * (size_t)Hn * En, w4);
    cvt_f32_bf16<<<(w4 + 255) / 256, 256, 0, stream>>>(Wv, wb + 2 * (size_t)Hn * En, w4);
  }

  // 2) QKV projections (V stored transposed)
  qkv_gemm<<<dim3(Mn / 16, 3), 256, 0, stream>>>(xb, wb, qb, kbuf, vtb);

  // 3) causal flash attention (128 query rows per block)
  flash_attn<<<dim3(Mn / 128), 256, 0, stream>>>(qb, kbuf, vtb, out);
}
